// MultiHeadSelfAttention_2095944040480
// MI455X (gfx1250) — matrix operand order repaired and checked by reading, not compiled, not run on hardware
//
#include <hip/hip_runtime.h>
#include <hip/hip_bf16.h>
#include <math.h>

// ---------------------------------------------------------------------------
// MI455X (gfx1250) multi-head self-attention forward.
// B=2, T=2048, C=1024, H=16, Dh=64.
// GEMMs: v_wmma_f32_16x16x32_bf16 with TDM (tensor_load_to_lds) B-tile staging.
// Attention: per-wave causal flash attention; Q pre-scaled by 1/sqrt(Dh)
// (exact in bf16), row-sum via ones-column WMMA, row-max via
// v_permlane16_b32 butterflies (no LDS in the softmax).
// ---------------------------------------------------------------------------

typedef __bf16 bf16x16 __attribute__((ext_vector_type(16)));
typedef __bf16 bf16x8  __attribute__((ext_vector_type(8)));
typedef float  f32x8   __attribute__((ext_vector_type(8)));
typedef unsigned int u32x4 __attribute__((ext_vector_type(4)));
typedef int    i32x4  __attribute__((ext_vector_type(4)));
typedef int    i32x8  __attribute__((ext_vector_type(8)));

__device__ __forceinline__ f32x8 wmma_bf16(bf16x16 a, bf16x16 b, f32x8 c) {
  return __builtin_amdgcn_wmma_f32_16x16x32_bf16(
      false, a, false, b, (short)0, c, false, false);
}

__device__ __forceinline__ bf16x16 join8(bf16x8 lo, bf16x8 hi) {
  return __builtin_shufflevector(lo, hi, 0,1,2,3,4,5,6,7,8,9,10,11,12,13,14,15);
}

__device__ __forceinline__ unsigned ldsOff(const void* p) {
  return (unsigned)(uintptr_t)p;   // low 32 bits of a generic LDS ptr = offset
}

// v_permlane16_b32 lane shuffle within each 16-lane row (VALU, no LDS pipe)
__device__ __forceinline__ float permx(float v, unsigned lo, unsigned hi) {
  unsigned u = __builtin_bit_cast(unsigned, v);
  unsigned r = __builtin_amdgcn_permlane16(u, u, lo, hi, false, false);
  return __builtin_bit_cast(float, r);
}

// max across the 16 lanes of a row via xor-butterfly permlane16s
__device__ __forceinline__ float rowmax16(float t) {
  t = fmaxf(t, permx(t, 0x67452301u, 0xEFCDAB89u));  // xor 1
  t = fmaxf(t, permx(t, 0x45670123u, 0xCDEF89ABu));  // xor 2
  t = fmaxf(t, permx(t, 0x01234567u, 0x89ABCDEFu));  // xor 4
  t = fmaxf(t, permx(t, 0xFEDCBA98u, 0x76543210u));  // xor 8
  return t;
}

// ---------------------------------------------------------------------------
// TDM: 64-row x 32-col bf16 tile (row stride K elems) global -> LDS, padding
// each 64B row by 16B (LDS row stride 80B: bank-spread, 16B-aligned frags).
// ---------------------------------------------------------------------------
__device__ __forceinline__ void tdm_load_tileB(const __bf16* gptr,
                                               unsigned lds_byte_off, int K) {
  const unsigned long long ga = (unsigned long long)(uintptr_t)gptr;
  u32x4 g0;
  g0.x = 1u;                                          // count=1 (valid user D#)
  g0.y = lds_byte_off;                                // lds_addr
  g0.z = (unsigned)(ga & 0xFFFFFFFFu);                // global_addr[31:0]
  g0.w = (unsigned)((ga >> 32) & 0x01FFFFFFu) | (2u << 30);  // [56:32] | type=2
  i32x8 g1;
  // data_size=1 (2B) | pad_enable | pad_interval=3 (16 DW) | pad_amount=3 (4 DW)
  g1[0] = (int)((1u << 16) | (1u << 20) | (3u << 22) | (3u << 25));
  g1[1] = (int)(((unsigned)K & 0xFFFFu) << 16);       // tensor_dim0[15:0]
  g1[2] = (int)(((unsigned)K >> 16) | (64u << 16));   // td0[31:16] | td1[15:0]
  g1[3] = (int)(32u << 16);                           // tile_dim0=32
  g1[4] = 64;                                         // tile_dim1=64
  g1[5] = K;                                          // tensor_dim0_stride
  g1[6] = 0;
  g1[7] = 0;
  const i32x4 z4 = {0, 0, 0, 0};
  const i32x8 z8 = {0, 0, 0, 0, 0, 0, 0, 0};
  __builtin_amdgcn_tensor_load_to_lds(g0, g1, z4, z4, z8, 0);
}

// ---------------------------------------------------------------------------
// Elementwise converters
// ---------------------------------------------------------------------------
__global__ void cvt_f32_to_bf16(const float* __restrict__ in,
                                __bf16* __restrict__ out, int n) {
  int i = blockIdx.x * blockDim.x + threadIdx.x;
  if (i < n) out[i] = (__bf16)in[i];
}

__global__ void cvt_transpose_bf16(const float* __restrict__ in,
                                   __bf16* __restrict__ out,
                                   int rows_in, int cols_in) {
  int i = blockIdx.x * blockDim.x + threadIdx.x;
  if (i < rows_in * cols_in) {
    int n = i / rows_in;
    int k = i - n * rows_in;
    out[i] = (__bf16)in[(size_t)k * cols_in + n];
  }
}

__global__ void build_vt(const __bf16* __restrict__ qkv,
                         __bf16* __restrict__ VT) {
  int i = blockIdx.x * blockDim.x + threadIdx.x;
  if (i >= (2 * 16 * 64 * 2048)) return;
  const int t  = i & 2047;
  const int dh = (i >> 11) & 63;
  const int h  = (i >> 17) & 15;
  const int b  = i >> 21;
  VT[i] = qkv[(size_t)(b * 2048 + t) * 3072 + 2048 + h * 64 + dh];
}

// ---------------------------------------------------------------------------
// GEMM: out(M,N) = A(M,K) @ BT(N,K)^T + bias.
// Block = 8 waves, 128(M) x 64(N).  TDM streams the shared B tile into LDS
// (double-buffered); A-frags straight from global.
// ---------------------------------------------------------------------------
template <bool F32OUT>
__global__ void __launch_bounds__(256)
gemm16_bf16(const __bf16* __restrict__ A, const __bf16* __restrict__ BT,
            const float* __restrict__ bias, float* __restrict__ outF,
            __bf16* __restrict__ outB, int M, int N, int K) {
  const int lane = threadIdx.x & 31;
  const int wIn  = threadIdx.x >> 5;
  const int tilesN = N >> 6;
  const int bm = blockIdx.x / tilesN;
  const int bn = blockIdx.x - bm * tilesN;
  const int m0 = (bm * 8 + wIn) << 4;
  const int n0 = bn << 6;
  if (m0 >= M) return;
  const int hh = lane >> 4, lm = lane & 15;

  __shared__ __bf16 bsh[2][64 * 40];

  const __bf16* gB = BT + (size_t)n0 * K;

  if (wIn == 0) {
    tdm_load_tileB(gB, ldsOff(&bsh[0][0]), K);
    __builtin_amdgcn_s_wait_tensorcnt(0);
  }
  __syncthreads();

  f32x8 acc0 = {}, acc1 = {}, acc2 = {}, acc3 = {};
  const __bf16* aRow = A + (size_t)(m0 + lm) * K + hh * 16;

  for (int k0 = 0; k0 < K; k0 += 32) {
    const int cur = (k0 >> 5) & 1;
    if (wIn == 0 && (k0 + 32) < K)
      tdm_load_tileB(gB + k0 + 32, ldsOff(&bsh[cur ^ 1][0]), K);

    bf16x16 av = join8(*(const bf16x8*)(aRow + k0),
                       *(const bf16x8*)(aRow + k0 + 8));
    const __bf16* bb = &bsh[cur][lm * 40 + hh * 16];
    bf16x16 b0 = join8(*(const bf16x8*)(bb +    0), *(const bf16x8*)(bb +    8));
    bf16x16 b1 = join8(*(const bf16x8*)(bb +  640), *(const bf16x8*)(bb +  648));
    bf16x16 b2 = join8(*(const bf16x8*)(bb + 1280), *(const bf16x8*)(bb + 1288));
    bf16x16 b3 = join8(*(const bf16x8*)(bb + 1920), *(const bf16x8*)(bb + 1928));
    acc0 = wmma_bf16(av, b0, acc0);
    acc1 = wmma_bf16(av, b1, acc1);
    acc2 = wmma_bf16(av, b2, acc2);
    acc3 = wmma_bf16(av, b3, acc3);
    __builtin_prefetch(aRow + k0 + 128, 0, 0);

    if (wIn == 0) __builtin_amdgcn_s_wait_tensorcnt(0);
    __syncthreads();
  }

#pragma unroll
  for (int r = 0; r < 8; ++r) {
    const size_t row = (size_t)(m0 + r + hh * 8);
    const int col0 = n0 + lm;
    const float v0 = acc0[r] + bias[col0];
    const float v1 = acc1[r] + bias[col0 + 16];
    const float v2 = acc2[r] + bias[col0 + 32];
    const float v3 = acc3[r] + bias[col0 + 48];
    if constexpr (F32OUT) {
      float* dst = outF + row * N + col0;
      dst[0] = v0; dst[16] = v1; dst[32] = v2; dst[48] = v3;
    } else {
      __bf16* dst = outB + row * N + col0;
      dst[0] = (__bf16)v0; dst[16] = (__bf16)v1;
      dst[32] = (__bf16)v2; dst[48] = (__bf16)v3;
    }
  }
}

// ---------------------------------------------------------------------------
// One 32-key flash-attention step.  MASKED only for the diagonal tile.
// Q is pre-scaled by 1/sqrt(Dh), so S comes out of the WMMA already scaled.
// Row sums accumulate on the matrix pipe via the ones-column B (bones).
// ---------------------------------------------------------------------------
template <bool MASKED>
__device__ __forceinline__ void attn_step(
    int k0, int q0, int hh, int lm, const __bf16* kbase, const __bf16* vbase,
    __bf16* P, const bf16x16& aq0, const bf16x16& aq1, const bf16x16& bones,
    float (&mrow)[8], f32x8& ol, f32x8& o0, f32x8& o1, f32x8& o2, f32x8& o3) {
  constexpr int T = 2048, C3 = 3072;
  const __bf16* k0row = kbase + (size_t)(k0 + lm) * C3;
  const __bf16* k1row = kbase + (size_t)(k0 + 16 + lm) * C3;
  f32x8 s0 = {}, s1 = {};
  s0 = wmma_bf16(aq0, *(const bf16x16*)(k0row), s0);
  s0 = wmma_bf16(aq1, *(const bf16x16*)(k0row + 32), s0);
  s1 = wmma_bf16(aq0, *(const bf16x16*)(k1row), s1);
  s1 = wmma_bf16(aq1, *(const bf16x16*)(k1row + 32), s1);

#pragma unroll
  for (int r = 0; r < 8; ++r) {
    float v0 = s0[r];
    float v1 = s1[r];
    if (MASKED) {
      const int q = q0 + r + hh * 8;
      v0 = (k0 + lm <= q)      ? v0 : -__builtin_inff();
      v1 = (k0 + 16 + lm <= q) ? v1 : -__builtin_inff();
    }
    const float t = rowmax16(fmaxf(v0, v1));
    const float nm = fmaxf(mrow[r], t);
    const float al = __expf(mrow[r] - nm);
    mrow[r] = nm;
    const float p0 = __expf(v0 - nm);
    const float p1 = __expf(v1 - nm);
    o0[r] *= al; o1[r] *= al; o2[r] *= al; o3[r] *= al; ol[r] *= al;
    P[((r + hh * 8) << 5) + lm]      = (__bf16)p0;
    P[((r + hh * 8) << 5) + 16 + lm] = (__bf16)p1;
  }

  // per-wave LDS RAW fence (no workgroup barrier: trip counts differ per wave)
  asm volatile("s_wait_dscnt 0" ::: "memory");

  const __bf16* prow = P + (lm << 5) + hh * 16;
  const bf16x16 ap = join8(*(const bf16x8*)(prow), *(const bf16x8*)(prow + 8));

  o0 = wmma_bf16(ap, *(const bf16x16*)(vbase + (size_t)(0 * 16) * T + k0), o0);
  o1 = wmma_bf16(ap, *(const bf16x16*)(vbase + (size_t)(1 * 16) * T + k0), o1);
  o2 = wmma_bf16(ap, *(const bf16x16*)(vbase + (size_t)(2 * 16) * T + k0), o2);
  o3 = wmma_bf16(ap, *(const bf16x16*)(vbase + (size_t)(3 * 16) * T + k0), o3);
  ol = wmma_bf16(ap, bones, ol);               // row sums of P (column 0)
}

// ---------------------------------------------------------------------------
// Causal flash attention: one wave per 16-query tile of one (b,h).
// ---------------------------------------------------------------------------
__global__ void __launch_bounds__(256)
flash_attn_bf16(const __bf16* __restrict__ qkv, const __bf16* __restrict__ VT,
                __bf16* __restrict__ attn) {
  constexpr int T = 2048, C3 = 3072, H = 16;
  const int lane = threadIdx.x & 31;
  const int wave = (int)((blockIdx.x * blockDim.x + threadIdx.x) >> 5);
  const int qt = wave & 127;
  const int h  = (wave >> 7) & 15;
  const int b  = wave >> 11;
  if (b >= 2) return;
  const int hh = lane >> 4, lm = lane & 15;
  const int q0 = qt << 4;

  __shared__ __bf16 plds[8 * 512];
  __bf16* P = plds + ((threadIdx.x >> 5) << 9);

  const __bf16* qrow = qkv + (size_t)(b * T + q0 + lm) * C3 + h * 64 + hh * 16;
  bf16x16 aq0 = join8(*(const bf16x8*)(qrow),
                      *(const bf16x8*)(qrow + 8));
  bf16x16 aq1 = join8(*(const bf16x8*)(qrow + 32),
                      *(const bf16x8*)(qrow + 40));
  // fold 1/sqrt(Dh)=0.125 into Q once: exact in bf16 (exponent shift only)
#pragma unroll
  for (int i = 0; i < 16; ++i) {
    aq0[i] = (__bf16)(0.125f * (float)aq0[i]);
    aq1[i] = (__bf16)(0.125f * (float)aq1[i]);
  }

  const __bf16* kbase = qkv + (size_t)b * T * C3 + 1024 + h * 64 + hh * 16;
  const __bf16* vbase = VT + (size_t)((b * H + h) * 64 + lm) * T + hh * 16;

  // ones-column B: B[k,n] = (n==0), i.e. lanes with lm==0 hold 1.0
  const __bf16 onev = (lm == 0) ? (__bf16)1.0f : (__bf16)0.0f;
  bf16x16 bones;
#pragma unroll
  for (int i = 0; i < 16; ++i) bones[i] = onev;

  float mrow[8];
#pragma unroll
  for (int r = 0; r < 8; ++r) mrow[r] = -__builtin_inff();
  f32x8 o0 = {}, o1 = {}, o2 = {}, o3 = {}, ol = {};

  const int kmask = q0 & ~31;   // only this tile straddles the diagonal
  for (int k0 = 0; k0 < kmask; k0 += 32)
    attn_step<false>(k0, q0, hh, lm, kbase, vbase, P, aq0, aq1, bones,
                     mrow, ol, o0, o1, o2, o3);
  attn_step<true>(kmask, q0, hh, lm, kbase, vbase, P, aq0, aq1, bones,
                  mrow, ol, o0, o1, o2, o3);

#pragma unroll
  for (int r = 0; r < 8; ++r) {
    // l lives in column 0 (lane 0 of each 16-lane row) -> broadcast
    const float l = permx(ol[r], 0u, 0u);
    const float inv = 1.0f / l;
    __bf16* dst = attn + (size_t)(b * T + q0 + r + hh * 8) * 1024 + h * 64 + lm;
    dst[0]  = (__bf16)(o0[r] * inv);
    dst[16] = (__bf16)(o1[r] * inv);
    dst[32] = (__bf16)(o2[r] * inv);
    dst[48] = (__bf16)(o3[r] * inv);
  }
}

// ---------------------------------------------------------------------------
// Host-side launch
// ---------------------------------------------------------------------------
extern "C" void kernel_launch(void* const* d_in, const int* in_sizes, int n_in,
                              void* d_out, int out_size, void* d_ws,
                              size_t ws_size, hipStream_t stream) {
  (void)in_sizes; (void)n_in; (void)out_size; (void)ws_size;
  const float* x     = (const float*)d_in[0];
  const float* W_qkv = (const float*)d_in[1];
  const float* b_qkv = (const float*)d_in[2];
  const float* W_out = (const float*)d_in[3];
  const float* b_out = (const float*)d_in[4];
  float* out = (float*)d_out;

  char* ws = (char*)d_ws;
  __bf16* X16    = (__bf16*)(ws);               //  8 MB  (4096,1024)
  __bf16* WqkvT  = (__bf16*)(ws +  8388608);    //  6 MB  (3072,1024)
  __bf16* WoutT  = (__bf16*)(ws + 14680064);    //  2 MB  (1024,1024)
  __bf16* qkv16  = (__bf16*)(ws + 16777216);    // 24 MB  (4096,3072)
  __bf16* VT16   = (__bf16*)(ws + 41943040);    //  8 MB  (2,16,64,2048)
  __bf16* attn16 = (__bf16*)(ws + 50331648);    //  8 MB  (4096,1024)

  cvt_f32_to_bf16<<<4194304 / 256, 256, 0, stream>>>(x, X16, 4194304);
  cvt_transpose_bf16<<<3145728 / 256, 256, 0, stream>>>(W_qkv, WqkvT, 1024, 3072);
  cvt_transpose_bf16<<<1048576 / 256, 256, 0, stream>>>(W_out, WoutT, 1024, 1024);

  gemm16_bf16<false><<<1536, 256, 0, stream>>>(X16, WqkvT, b_qkv, nullptr,
                                               qkv16, 4096, 3072, 1024);

  build_vt<<<16384, 256, 0, stream>>>(qkv16, VT16);

  flash_attn_bf16<<<512, 256, 0, stream>>>(qkv16, VT16, attn16);

  gemm16_bf16<true><<<512, 256, 0, stream>>>(attn16, WoutT, b_out, out,
                                             nullptr, 4096, 1024, 1024);
}
